// mv3Dunet_down_text_cmsa_org_small_15633680958015
// MI455X (gfx1250) — matrix operand order repaired and checked by reading, not compiled, not run on hardware
//
#include <hip/hip_runtime.h>

// ---------------------------------------------------------------------------
// Types for CDNA5 WMMA (wave32, 16x16x32 bf16 -> f32 accumulate)
// ---------------------------------------------------------------------------
typedef __attribute__((ext_vector_type(16))) __bf16          v16bf;
typedef __attribute__((ext_vector_type(16))) unsigned short  v16u;
typedef __attribute__((ext_vector_type(8)))  float           v8f;

__device__ __forceinline__ unsigned short f2bf_bits(float f) {
  unsigned u = __builtin_bit_cast(unsigned, f);
  u += 0x7FFFu + ((u >> 16) & 1u);     // round-to-nearest-even
  return (unsigned short)(u >> 16);
}

// A-fragment K index for vgpr-pair v (0..7), lane half (0/1): lane holds M row.
__device__ __forceinline__ constexpr int a_kidx(int v, int hlf) {
  return ((v & 4) << 2) + ((v & 3) << 1) + (hlf << 3);
}
// B-fragment K index: lane holds N column.
__device__ __forceinline__ constexpr int b_kidx(int v, int hlf) {
  return ((v & 4) << 2) + ((v & 3) << 1) + (hlf << 3);
}

// ---------------------------------------------------------------------------
// Generic strided batched GEMM:  C[b] = act( A[b] x B[b] + bias ) + res[b]
//   A element (m,k) = A[b*sAb + m*sAm + k*sAk]
//   B element (k,n) = B[b*sBb + k*sBk + n*sBn]
//   C row-major [M,N], batch stride sCb.  biasMode: 0 none, 1 per-col, 2 per-row
//   relu applied BEFORE residual add (matches reference relu(lin(x)) + t).
// One wave per 16x16 tile.  AKC/BKC: compile-time K-strides (0 = runtime).
// Out-of-tile m/n lanes load from clamped addresses (results never stored),
// so the K main loop is completely branch- and select-free.
// ---------------------------------------------------------------------------
template<int AKC, int BKC>
__global__ __launch_bounds__(32) void gemm_wmma_kernel(
    const float* __restrict__ A, int sAm, int sAkArg, long sAb,
    const float* __restrict__ Bm, int sBkArg, int sBn, long sBb,
    float* __restrict__ C, long sCb,
    const float* __restrict__ bias, int biasMode,
    const float* __restrict__ res, long sRb, int relu,
    int M, int N, int K, int MT, int NT)
{
  const int sAk = AKC ? AKC : sAkArg;
  const int sBk = BKC ? BKC : sBkArg;

  int wid = blockIdx.x;
  int nt = wid % NT; wid /= NT;
  int mt = wid % MT; wid /= MT;
  int bb = wid;

  int lane = threadIdx.x;
  int hlf  = lane >> 4;
  int lm   = lane & 15;

  int m = mt * 16 + lm;
  int n = nt * 16 + lm;
  int mc = (m < M) ? m : (M - 1);      // clamped (garbage rows never stored)
  int nc = (n < N) ? n : (N - 1);

  const float* ap = A  + (size_t)bb * sAb + (size_t)mc * sAm;
  const float* bp = Bm + (size_t)bb * sBb + (size_t)nc * sBn;

  // loop-invariant per-lane element offsets (constants when AKC/BKC != 0)
  int aoff[16], boff[16];
#pragma unroll
  for (int v = 0; v < 8; ++v) {
    aoff[2 * v]     = a_kidx(v, hlf) * sAk;
    aoff[2 * v + 1] = (a_kidx(v, hlf) + 1) * sAk;
    boff[2 * v]     = b_kidx(v, hlf) * sBk;
    boff[2 * v + 1] = (b_kidx(v, hlf) + 1) * sBk;
  }

  v8f acc;
#pragma unroll
  for (int i = 0; i < 8; ++i) acc[i] = 0.0f;

  int kmain = K & ~31;
  for (int k0 = 0; k0 < kmain; k0 += 32) {
    v16u au, bu;
#pragma unroll
    for (int i = 0; i < 16; ++i) {
      au[i] = f2bf_bits(ap[aoff[i]]);
      bu[i] = f2bf_bits(bp[boff[i]]);
    }
    v16bf af = __builtin_bit_cast(v16bf, au);
    v16bf bf = __builtin_bit_cast(v16bf, bu);
    acc = __builtin_amdgcn_wmma_f32_16x16x32_bf16(false, af, false, bf, (short)0, acc, false, false);
    ap += (size_t)32 * sAk;
    bp += (size_t)32 * sBk;
  }
  if (kmain < K) {                      // single tail step, branchless selects
    int rem = K - kmain;
    v16u au, bu;
#pragma unroll
    for (int v = 0; v < 8; ++v) {
      int ka0 = a_kidx(v, hlf), ka1 = ka0 + 1;
      int kb0 = b_kidx(v, hlf), kb1 = kb0 + 1;
      unsigned short ta0 = f2bf_bits(ap[(ka0 < rem ? ka0 : 0) * sAk]);
      unsigned short ta1 = f2bf_bits(ap[(ka1 < rem ? ka1 : 0) * sAk]);
      unsigned short tb0 = f2bf_bits(bp[(kb0 < rem ? kb0 : 0) * sBk]);
      unsigned short tb1 = f2bf_bits(bp[(kb1 < rem ? kb1 : 0) * sBk]);
      au[2 * v]     = (ka0 < rem) ? ta0 : (unsigned short)0;
      au[2 * v + 1] = (ka1 < rem) ? ta1 : (unsigned short)0;
      bu[2 * v]     = (kb0 < rem) ? tb0 : (unsigned short)0;
      bu[2 * v + 1] = (kb1 < rem) ? tb1 : (unsigned short)0;
    }
    v16bf af = __builtin_bit_cast(v16bf, au);
    v16bf bf = __builtin_bit_cast(v16bf, bu);
    acc = __builtin_amdgcn_wmma_f32_16x16x32_bf16(false, af, false, bf, (short)0, acc, false, false);
  }

  if (n >= N) return;
  int gm0 = mt * 16 + (hlf << 3);
  float bcol = (biasMode == 1) ? bias[n] : 0.0f;
#pragma unroll
  for (int r = 0; r < 8; ++r) {
    int mm = gm0 + r;
    if (mm < M) {
      float v = acc[r] + bcol;
      if (biasMode == 2) v += bias[mm];
      if (relu) v = fmaxf(v, 0.0f);
      if (res) v += res[(size_t)bb * sRb + (size_t)mm * N + n];
      C[(size_t)bb * sCb + (size_t)mm * N + n] = v;
    }
  }
}

// ---------------------------------------------------------------------------
// Implicit-GEMM 3x3x3 'same' conv, NCDHW, D=4 H=7 W=7 (196 spatial).
// out[b, o, s] = bias[o] + sum_{c,tap} w[o,c,tap] * x[b, c, s(+tap)] (0-pad)
// One wave per (batch, 16-outchan tile, 16-spatial tile). Cin % 32 == 0.
// All loads unconditional (boundary lanes load from tap-offset 0 and are
// masked after conversion with a cndmask, never a branch).
// ---------------------------------------------------------------------------
__global__ __launch_bounds__(32) void conv3d_wmma_kernel(
    const float* __restrict__ X, const float* __restrict__ Wt,
    const float* __restrict__ bias, float* __restrict__ Y,
    int Cin, int Cout, int MT, int NT)
{
  int wid = blockIdx.x;
  int nt = wid % NT; wid /= NT;
  int mt = wid % MT; wid /= MT;
  int bb = wid;

  int lane = threadIdx.x;
  int hlf  = lane >> 4;
  int lm   = lane & 15;

  int s  = nt * 16 + lm;                 // spatial index (B-fragment lane col)
  int sd = s / 49, sr = s % 49, sh = sr / 7, sw = sr % 7;
  int m  = mt * 16 + lm;                 // out-channel (A-fragment lane row)

  const float* xb0  = X  + (size_t)bb * Cin * 196;
  const float* wrow = Wt + (size_t)m * Cin * 27;

  // compile-time per-lane element offsets (fold into load immediates)
  int woff[16], xoff[16];
#pragma unroll
  for (int v = 0; v < 8; ++v) {
    woff[2 * v]     = a_kidx(v, hlf) * 27;
    woff[2 * v + 1] = (a_kidx(v, hlf) + 1) * 27;
    xoff[2 * v]     = b_kidx(v, hlf) * 196;
    xoff[2 * v + 1] = (b_kidx(v, hlf) + 1) * 196;
  }

  v8f acc;
#pragma unroll
  for (int i = 0; i < 8; ++i) acc[i] = 0.0f;

  for (int tap = 0; tap < 27; ++tap) {
    int kd = tap / 9, kh = (tap % 9) / 3, kw = tap % 3;
    int dd = sd + kd - 1, hh = sh + kh - 1, ww = sw + kw - 1;
    bool valid = (s < 196) && ((unsigned)dd < 4u) && ((unsigned)hh < 7u) && ((unsigned)ww < 7u);
    const float* xp = xb0 + (valid ? (dd * 49 + hh * 7 + ww) : 0);
    const float* wp = wrow + tap;
    __builtin_prefetch(wp, 0, 1);        // global_prefetch_b8 on gfx1250

    for (int c0 = 0; c0 < Cin; c0 += 32) {
      v16u au, bu;
#pragma unroll
      for (int i = 0; i < 16; ++i) {
        au[i] = f2bf_bits(wp[woff[i]]);
        unsigned short t = f2bf_bits(xp[xoff[i]]);   // load always issued
        bu[i] = valid ? t : (unsigned short)0;       // cndmask, no branch
      }
      v16bf af = __builtin_bit_cast(v16bf, au);
      v16bf bf = __builtin_bit_cast(v16bf, bu);
      acc = __builtin_amdgcn_wmma_f32_16x16x32_bf16(false, af, false, bf, (short)0, acc, false, false);
      wp += 32 * 27;
      xp += 32 * 196;
    }
  }

  int gn = nt * 16 + lm;
  if (gn >= 196) return;
  int gm0 = mt * 16 + (hlf << 3);
#pragma unroll
  for (int r = 0; r < 8; ++r) {
    int mm = gm0 + r;
    Y[(size_t)bb * Cout * 196 + (size_t)mm * 196 + gn] = acc[r] + bias[mm];
  }
}

// ---------------------------------------------------------------------------
// BatchNorm (training mode, biased var) over (batch, spatial) per channel.
// ---------------------------------------------------------------------------
__global__ void bn_stats_kernel(const float* __restrict__ X, float* __restrict__ stats,
                                int C, int NS)
{
  __shared__ float s1[256], s2[256];
  int c = blockIdx.x, t = threadIdx.x;
  float a = 0.f, b = 0.f;
  for (int i = t; i < NS; i += 256) {
    int bb = i / 196, sp = i % 196;
    float v = X[((size_t)bb * C + c) * 196 + sp];
    a += v; b += v * v;
  }
  s1[t] = a; s2[t] = b; __syncthreads();
  for (int off = 128; off > 0; off >>= 1) {
    if (t < off) { s1[t] += s1[t + off]; s2[t] += s2[t + off]; }
    __syncthreads();
  }
  if (t == 0) {
    float mean = s1[0] / (float)NS;
    stats[2 * c]     = mean;
    stats[2 * c + 1] = s2[0] / (float)NS - mean * mean;
  }
}

__global__ void bn_apply_relu_kernel(float* __restrict__ X, const float* __restrict__ stats,
                                     const float* __restrict__ g, const float* __restrict__ be,
                                     int C, int total)
{
  int idx = blockIdx.x * 256 + threadIdx.x;
  if (idx >= total) return;
  int c = (idx / 196) % C;
  float v = (X[idx] - stats[2 * c]) * rsqrtf(stats[2 * c + 1] + 1e-5f) * g[c] + be[c];
  X[idx] = fmaxf(v, 0.0f);
}

// ---------------------------------------------------------------------------
// catnum: t = Linear(in) -> BN over batch -> SiLU.  out [32,196]
// ---------------------------------------------------------------------------
__global__ void catnum_kernel(const float* __restrict__ in, const float* __restrict__ w,
                              const float* __restrict__ b_, const float* __restrict__ g,
                              const float* __restrict__ be, float* __restrict__ out, int K)
{
  int j = threadIdx.x;
  if (j >= 196) return;
  float t[32];
  float mean = 0.f;
  for (int bb = 0; bb < 32; ++bb) {
    float a = b_[j];
    for (int k = 0; k < K; ++k) a += in[bb * K + k] * w[j * K + k];
    t[bb] = a; mean += a;
  }
  mean *= (1.0f / 32.0f);
  float var = 0.f;
  for (int bb = 0; bb < 32; ++bb) { float d = t[bb] - mean; var += d * d; }
  var *= (1.0f / 32.0f);
  float inv = rsqrtf(var + 1e-5f);
  for (int bb = 0; bb < 32; ++bb) {
    float y = (t[bb] - mean) * inv * g[j] + be[j];
    y = y / (1.0f + expf(-y));                     // SiLU
    out[bb * 196 + j] = y;
  }
}

__global__ void broadcast_text_kernel(const float* __restrict__ toh, const float* __restrict__ tnm,
                                      float* __restrict__ text)
{
  int idx = blockIdx.x * 256 + threadIdx.x;
  if (idx >= 32 * 45 * 196) return;
  int b = idx / (45 * 196), rem = idx % (45 * 196);
  int ch = rem / 196, sp = rem % 196;
  text[idx] = (ch < 30) ? toh[b * 196 + sp] : tnm[b * 196 + sp];
}

// dist[b,i,j] = exp(-sum_k |f[b,i,k]-f[b,j,k]|)
__global__ void l1dist_kernel(const float* __restrict__ F, float* __restrict__ D, int C)
{
  __shared__ float fi[196];
  int bi = blockIdx.x;
  int b = bi / C, i = bi % C;
  const float* frow = F + ((size_t)b * C + i) * 196;
  for (int t = threadIdx.x; t < 196; t += blockDim.x) fi[t] = frow[t];
  __syncthreads();
  for (int j = threadIdx.x; j < C; j += blockDim.x) {
    const float* fj = F + ((size_t)b * C + j) * 196;
    float s = 0.f;
    for (int k = 0; k < 196; ++k) s += fabsf(fi[k] - fj[k]);
    D[((size_t)b * C + i) * C + j] = expf(-s);
  }
}

__global__ void layernorm_kernel(const float* __restrict__ X, float* __restrict__ Y,
                                 const float* __restrict__ g, const float* __restrict__ be)
{
  __shared__ float s1[256], s2[256];
  int row = blockIdx.x, t = threadIdx.x;
  const float* xr = X + (size_t)row * 196;
  float v = (t < 196) ? xr[t] : 0.f;
  s1[t] = v; s2[t] = v * v; __syncthreads();
  for (int off = 128; off > 0; off >>= 1) {
    if (t < off) { s1[t] += s1[t + off]; s2[t] += s2[t + off]; }
    __syncthreads();
  }
  float mean = s1[0] * (1.0f / 196.0f);
  float var  = s2[0] * (1.0f / 196.0f) - mean * mean;
  float inv  = rsqrtf(var + 1e-6f);
  if (t < 196) Y[(size_t)row * 196 + t] = (v - mean) * inv * g[t] + be[t];
}

__global__ void softmax_rows_kernel(float* __restrict__ S, int L)
{
  __shared__ float red[256];
  float* row = S + (size_t)blockIdx.x * L;
  int t = threadIdx.x;
  float v = (t < L) ? row[t] : -3.0e38f;
  red[t] = v; __syncthreads();
  for (int off = 128; off > 0; off >>= 1) {
    if (t < off) red[t] = fmaxf(red[t], red[t + off]);
    __syncthreads();
  }
  float mx = red[0]; __syncthreads();
  float e = (t < L) ? expf(v - mx) : 0.f;
  red[t] = e; __syncthreads();
  for (int off = 128; off > 0; off >>= 1) {
    if (t < off) red[t] += red[t + off];
    __syncthreads();
  }
  float s = red[0];
  if (t < L) row[t] = e / s;
}

// dst[b, dstOff + r*196 + s] = A[b,r,s] + B[b,r,s]
__global__ void add_concat_kernel(const float* __restrict__ A, const float* __restrict__ Bp,
                                  float* __restrict__ dst, long srcBS, long dstBS, long dstOff,
                                  int total)
{
  int idx = blockIdx.x * 256 + threadIdx.x;
  if (idx >= total) return;
  int b = idx / (int)srcBS; long rem = idx % (int)srcBS;
  dst[(size_t)b * dstBS + dstOff + rem] = A[(size_t)b * srcBS + rem] + Bp[(size_t)b * srcBS + rem];
}

__global__ void rowsum_kernel(const float* __restrict__ X, float* __restrict__ out, int rows, int L)
{
  int r = blockIdx.x * 256 + threadIdx.x;
  if (r >= rows) return;
  float s = 0.f;
  for (int i = 0; i < L; ++i) s += X[(size_t)r * L + i];
  out[r] = s;
}

// final: concat([sum(t_oh) dup x30, sum(t_nm) dup x15, pcm(173)]) @ cls_w^T + cls_b
__global__ void classifier_kernel(const float* __restrict__ toh, const float* __restrict__ tnm,
                                  const float* __restrict__ pcm, const float* __restrict__ w,
                                  const float* __restrict__ bb, float* __restrict__ out)
{
  int id = threadIdx.x;
  if (id >= 64) return;
  int b = id >> 1, k = id & 1;
  float soh = 0.f, snm = 0.f;
  for (int i = 0; i < 196; ++i) { soh += toh[b * 196 + i]; snm += tnm[b * 196 + i]; }
  const float* wr = w + k * 218;
  float acc = bb[k];
  float w0 = 0.f; for (int c = 0;  c < 30; ++c) w0 += wr[c];       acc += soh * w0;
  float w1 = 0.f; for (int c = 30; c < 45; ++c) w1 += wr[c];       acc += snm * w1;
  for (int c = 0; c < 173; ++c) acc += pcm[b * 173 + c] * wr[45 + c];
  out[b * 2 + k] = acc;
}

// ---------------------------------------------------------------------------
// Host orchestration
// ---------------------------------------------------------------------------
extern "C" void kernel_launch(void* const* d_in, const int* in_sizes, int n_in,
                              void* d_out, int out_size, void* d_ws, size_t ws_size,
                              hipStream_t stream) {
  (void)in_sizes; (void)n_in; (void)out_size; (void)ws_size;
  const float* x      = (const float*)d_in[0];
  const float* oneHot = (const float*)d_in[1];
  const float* numv   = (const float*)d_in[2];
  const float* c1_w1  = (const float*)d_in[3];
  const float* c1_b1  = (const float*)d_in[4];
  const float* c1_g1  = (const float*)d_in[5];
  const float* c1_be1 = (const float*)d_in[6];
  const float* c1_w2  = (const float*)d_in[7];
  const float* c1_b2  = (const float*)d_in[8];
  const float* c1_g2  = (const float*)d_in[9];
  const float* c1_be2 = (const float*)d_in[10];
  const float* oh_w   = (const float*)d_in[11];
  const float* oh_b   = (const float*)d_in[12];
  const float* oh_g   = (const float*)d_in[13];
  const float* oh_be  = (const float*)d_in[14];
  const float* nm_w   = (const float*)d_in[15];
  const float* nm_b   = (const float*)d_in[16];
  const float* nm_g   = (const float*)d_in[17];
  const float* nm_be  = (const float*)d_in[18];
  const float* gm1_w  = (const float*)d_in[19];
  const float* gm1_b  = (const float*)d_in[20];
  const float* gm2_w  = (const float*)d_in[21];
  const float* gm2_b  = (const float*)d_in[22];
  const float* ln1_g  = (const float*)d_in[23];
  const float* ln1_b  = (const float*)d_in[24];
  const float* ln2_g  = (const float*)d_in[25];
  const float* ln2_b  = (const float*)d_in[26];
  const float* e1_w   = (const float*)d_in[27];
  const float* e1_b   = (const float*)d_in[28];
  const float* e2_w   = (const float*)d_in[29];
  const float* e2_b   = (const float*)d_in[30];
  const float* f1_w   = (const float*)d_in[31];
  const float* f1_b   = (const float*)d_in[32];
  const float* f2_w   = (const float*)d_in[33];
  const float* f2_b   = (const float*)d_in[34];
  const float* f3_w   = (const float*)d_in[35];
  const float* f3_b   = (const float*)d_in[36];
  const float* f4_w   = (const float*)d_in[37];
  const float* f4_b   = (const float*)d_in[38];
  const float* cm_w   = (const float*)d_in[39];
  const float* cm_b   = (const float*)d_in[40];
  const float* cmf_w  = (const float*)d_in[41];
  const float* cmf_b  = (const float*)d_in[42];
  const float* cls_w  = (const float*)d_in[43];
  const float* cls_b  = (const float*)d_in[44];

  float* W = (float*)d_ws;
  // workspace offsets (floats)
  const size_t OFF_H       = 0;         // [32,128,196]  (reused later: img_fea)
  const size_t OFF_CONV5   = 802816;    // [32,128,196]
  const size_t OFF_STATS   = 1605632;   // [128,2]
  const size_t OFF_TOH     = 1605888;   // [32,196]
  const size_t OFF_TNM     = 1612160;   // [32,196]
  const size_t OFF_TEXT    = 1618432;   // [32,45,196]
  const size_t OFF_DIST128 = 1900672;   // [32,128,128]  (reused: s_ii)
  const size_t OFF_GBUF    = 2424960;   // [32,128,196]  (reused: it_fea)
  const size_t OFF_CONV5G  = 3227776;   // [32,128,196]
  const size_t OFF_DIST45  = 4030592;   // [32,45,45]    (reused: s_tt)
  const size_t OFF_TEXTG   = 4095392;   // [32,45,196]
  const size_t OFF_LNIMG   = 4377632;
  const size_t OFF_LNTXT   = 5180448;
  const size_t OFF_IQ      = 5462688;
  const size_t OFF_TQ      = 6265504;
  const size_t OFF_SIT     = 6547744;   // [32,128,45]
  const size_t OFF_STI     = 6732064;   // [32,45,128]
  const size_t OFF_TEXTFEA = 6916384;
  const size_t OFF_TIFEA   = 7198624;
  const size_t OFF_CI      = 7480864;
  const size_t OFF_CT      = 8283680;
  const size_t OFF_FEAT    = 8565920;   // [32,173,196]
  const size_t OFF_QBUF    = 9650656;
  const size_t OFF_SCM     = 10735392;  // [32,196,196]
  const size_t OFF_FEA2    = 11964704;
  const size_t OFF_CMSA    = 13049440;
  const size_t OFF_PCM     = 14134176;  // [32,173]
  const size_t OFF_IMGFEA  = OFF_H;
  const size_t OFF_ITFEA   = OFF_GBUF;
  const size_t OFF_SII     = OFF_DIST128;
  const size_t OFF_STT     = OFF_DIST45;

  auto gemm = [&](const float* A, int sAm, int sAk, long sAb,
                  const float* Bp, int sBk, int sBn, long sBb,
                  float* Cp, long sCb,
                  const float* bias, int biasMode,
                  const float* res, long sRb, int relu,
                  int M, int N, int K) {
    int MT = (M + 15) >> 4, NT = (N + 15) >> 4;
    dim3 grid((unsigned)(32 * MT * NT)), blk(32);
    if (sAk == 1 && sBk == 1)
      gemm_wmma_kernel<1, 1><<<grid, blk, 0, stream>>>(A, sAm, sAk, sAb, Bp, sBk, sBn, sBb,
          Cp, sCb, bias, biasMode, res, sRb, relu, M, N, K, MT, NT);
    else if (sAk == 1 && sBk == 196)
      gemm_wmma_kernel<1, 196><<<grid, blk, 0, stream>>>(A, sAm, sAk, sAb, Bp, sBk, sBn, sBb,
          Cp, sCb, bias, biasMode, res, sRb, relu, M, N, K, MT, NT);
    else if (sAk == 196 && sBk == 196)
      gemm_wmma_kernel<196, 196><<<grid, blk, 0, stream>>>(A, sAm, sAk, sAb, Bp, sBk, sBn, sBb,
          Cp, sCb, bias, biasMode, res, sRb, relu, M, N, K, MT, NT);
    else
      gemm_wmma_kernel<0, 0><<<grid, blk, 0, stream>>>(A, sAm, sAk, sAb, Bp, sBk, sBn, sBb,
          Cp, sCb, bias, biasMode, res, sRb, relu, M, N, K, MT, NT);
  };

  // ---- stem: conv1 -> BN+ReLU -> conv2 -> BN+ReLU ------------------------
  conv3d_wmma_kernel<<<dim3(32 * 8 * 13), dim3(32), 0, stream>>>(x, c1_w1, c1_b1, W + OFF_H, 320, 128, 8, 13);
  bn_stats_kernel<<<128, 256, 0, stream>>>(W + OFF_H, W + OFF_STATS, 128, 32 * 196);
  bn_apply_relu_kernel<<<(32 * 128 * 196 + 255) / 256, 256, 0, stream>>>(W + OFF_H, W + OFF_STATS, c1_g1, c1_be1, 128, 32 * 128 * 196);
  conv3d_wmma_kernel<<<dim3(32 * 8 * 13), dim3(32), 0, stream>>>(W + OFF_H, c1_w2, c1_b2, W + OFF_CONV5, 128, 128, 8, 13);
  bn_stats_kernel<<<128, 256, 0, stream>>>(W + OFF_CONV5, W + OFF_STATS, 128, 32 * 196);
  bn_apply_relu_kernel<<<(32 * 128 * 196 + 255) / 256, 256, 0, stream>>>(W + OFF_CONV5, W + OFF_STATS, c1_g2, c1_be2, 128, 32 * 128 * 196);

  // ---- text branch -------------------------------------------------------
  catnum_kernel<<<1, 224, 0, stream>>>(oneHot, oh_w, oh_b, oh_g, oh_be, W + OFF_TOH, 24);
  catnum_kernel<<<1, 224, 0, stream>>>(numv,   nm_w, nm_b, nm_g, nm_be, W + OFF_TNM, 11);
  broadcast_text_kernel<<<(32 * 45 * 196 + 255) / 256, 256, 0, stream>>>(W + OFF_TOH, W + OFF_TNM, W + OFF_TEXT);

  // ---- GCN on conv5 (C=128) ---------------------------------------------
  l1dist_kernel<<<32 * 128, 128, 0, stream>>>(W + OFF_CONV5, W + OFF_DIST128, 128);
  gemm(W + OFF_DIST128, 128, 1, 16384, W + OFF_CONV5, 196, 1, 25088,
       W + OFF_GBUF, 25088, nullptr, 0, nullptr, 0, 0, 128, 196, 128);
  gemm(W + OFF_GBUF, 196, 1, 25088, gm1_w, 1, 196, 0,
       W + OFF_CONV5G, 25088, gm1_b, 1, W + OFF_CONV5, 25088, 1, 128, 196, 196);

  // ---- GCN on text (C=45) -----------------------------------------------
  l1dist_kernel<<<32 * 45, 128, 0, stream>>>(W + OFF_TEXT, W + OFF_DIST45, 45);
  gemm(W + OFF_DIST45, 45, 1, 2025, W + OFF_TEXT, 196, 1, 8820,
       W + OFF_GBUF, 8820, nullptr, 0, nullptr, 0, 0, 45, 196, 45);
  gemm(W + OFF_GBUF, 196, 1, 8820, gm2_w, 1, 196, 0,
       W + OFF_TEXTG, 8820, gm2_b, 1, W + OFF_TEXT, 8820, 1, 45, 196, 196);

  // ---- LayerNorm + cross-attention --------------------------------------
  layernorm_kernel<<<32 * 128, 256, 0, stream>>>(W + OFF_CONV5G, W + OFF_LNIMG, ln1_g, ln1_b);
  layernorm_kernel<<<32 * 45,  256, 0, stream>>>(W + OFF_TEXTG,  W + OFF_LNTXT, ln2_g, ln2_b);

  gemm(W + OFF_LNIMG, 196, 1, 25088, e1_w, 1, 196, 0, W + OFF_IQ, 25088, e1_b, 1, nullptr, 0, 1, 128, 196, 196);
  gemm(W + OFF_LNTXT, 196, 1, 8820,  e2_w, 1, 196, 0, W + OFF_TQ, 8820,  e2_b, 1, nullptr, 0, 1, 45,  196, 196);

  gemm(W + OFF_IQ, 196, 1, 25088, W + OFF_IQ, 1, 196, 25088, W + OFF_SII, 16384, nullptr, 0, nullptr, 0, 0, 128, 128, 196);
  gemm(W + OFF_TQ, 196, 1, 8820,  W + OFF_TQ, 1, 196, 8820,  W + OFF_STT, 2025,  nullptr, 0, nullptr, 0, 0, 45,  45,  196);
  gemm(W + OFF_IQ, 196, 1, 25088, W + OFF_TQ, 1, 196, 8820,  W + OFF_SIT, 5760,  nullptr, 0, nullptr, 0, 0, 128, 45,  196);
  gemm(W + OFF_TQ, 196, 1, 8820,  W + OFF_IQ, 1, 196, 25088, W + OFF_STI, 5760,  nullptr, 0, nullptr, 0, 0, 45,  128, 196);

  softmax_rows_kernel<<<32 * 128, 256, 0, stream>>>(W + OFF_SII, 128);
  softmax_rows_kernel<<<32 * 45,  256, 0, stream>>>(W + OFF_STT, 45);
  softmax_rows_kernel<<<32 * 128, 256, 0, stream>>>(W + OFF_SIT, 45);
  softmax_rows_kernel<<<32 * 45,  256, 0, stream>>>(W + OFF_STI, 128);

  gemm(W + OFF_SII, 128, 1, 16384, W + OFF_IQ, 196, 1, 25088, W + OFF_IMGFEA,  25088, nullptr, 0, nullptr, 0, 0, 128, 196, 128);
  gemm(W + OFF_STT, 45,  1, 2025,  W + OFF_TQ, 196, 1, 8820,  W + OFF_TEXTFEA, 8820,  nullptr, 0, nullptr, 0, 0, 45,  196, 45);
  gemm(W + OFF_SIT, 45,  1, 5760,  W + OFF_TQ, 196, 1, 8820,  W + OFF_ITFEA,   25088, nullptr, 0, nullptr, 0, 0, 128, 196, 45);
  gemm(W + OFF_STI, 128, 1, 5760,  W + OFF_IQ, 196, 1, 25088, W + OFF_TIFEA,   8820,  nullptr, 0, nullptr, 0, 0, 45,  196, 128);

  // ci = img_fea@f1^T + f1_b + it_fea@f3^T + f3_b + ln_img
  gemm(W + OFF_IMGFEA, 196, 1, 25088, f1_w, 1, 196, 0, W + OFF_CI, 25088, f1_b, 1, W + OFF_LNIMG, 25088, 0, 128, 196, 196);
  gemm(W + OFF_ITFEA,  196, 1, 25088, f3_w, 1, 196, 0, W + OFF_CI, 25088, f3_b, 1, W + OFF_CI,    25088, 0, 128, 196, 196);
  // ct = text_fea@f2^T + f2_b + ti_fea@f4^T + f4_b + ln_txt
  gemm(W + OFF_TEXTFEA, 196, 1, 8820, f2_w, 1, 196, 0, W + OFF_CT, 8820, f2_b, 1, W + OFF_LNTXT, 8820, 0, 45, 196, 196);
  gemm(W + OFF_TIFEA,   196, 1, 8820, f4_w, 1, 196, 0, W + OFF_CT, 8820, f4_b, 1, W + OFF_CT,    8820, 0, 45, 196, 196);

  // feat = concat(ci + conv5g, ct + textg)
  add_concat_kernel<<<(32 * 128 * 196 + 255) / 256, 256, 0, stream>>>(W + OFF_CI, W + OFF_CONV5G, W + OFF_FEAT, 25088, 33908, 0, 32 * 128 * 196);
  add_concat_kernel<<<(32 * 45 * 196 + 255) / 256, 256, 0, stream>>>(W + OFF_CT, W + OFF_TEXTG, W + OFF_FEAT, 8820, 33908, 25088, 32 * 45 * 196);

  // ---- CMSA --------------------------------------------------------------
  gemm(cm_w, 173, 1, 0, W + OFF_FEAT, 196, 1, 33908, W + OFF_QBUF, 33908, cm_b, 2, nullptr, 0, 1, 173, 196, 173);
  gemm(W + OFF_QBUF, 1, 196, 33908, W + OFF_QBUF, 196, 1, 33908, W + OFF_SCM, 38416, nullptr, 0, nullptr, 0, 0, 196, 196, 173);
  softmax_rows_kernel<<<32 * 196, 256, 0, stream>>>(W + OFF_SCM, 196);
  gemm(W + OFF_QBUF, 196, 1, 33908, W + OFF_SCM, 1, 196, 38416, W + OFF_FEA2, 33908, nullptr, 0, nullptr, 0, 0, 173, 196, 196);
  gemm(cmf_w, 173, 1, 0, W + OFF_FEA2, 196, 1, 33908, W + OFF_CMSA, 33908, cmf_b, 2, W + OFF_FEAT, 33908, 1, 173, 196, 173);

  // ---- pooled sums + classifier -----------------------------------------
  rowsum_kernel<<<(32 * 173 + 255) / 256, 256, 0, stream>>>(W + OFF_CMSA, W + OFF_PCM, 32 * 173, 196);
  classifier_kernel<<<1, 64, 0, stream>>>(W + OFF_TOH, W + OFF_TNM, W + OFF_PCM, cls_w, cls_b, (float*)d_out);
}
